// BiLSTM_CRF_56109452755670
// MI455X (gfx1250) — matrix operand order repaired and checked by reading, compiled, not run on hardware
//
#include <hip/hip_runtime.h>
#include <hip/hip_bf16.h>

// BiLSTM-CRF for gfx1250: V=50000 E=256 HD=512 (H=256/dir) B=32 S=256 T=24
//
// Pipeline:
//  1) k_gather : embedding gather -> X16 [S*B, 256] f16
//  2) k_prep   : Wih/Whh (both dirs) -> f16 [2048,256]; bias sums -> bsum[2048]
//  3) k_in_gemm: P = X16 @ W^T + bsum (v_wmma_f32_16x16x32_f16), 4 N-tiles/wave,
//                stored in WMMA-tile-native layout for the recurrence consumer
//  4) k_lstm   : 2 persistent workgroups (fwd/bwd), 32 waves each; per step
//                gates = P[s-tiles] + h16 @ Whh^T via WMMA, i/f/g/o per wave in regs
//  5) k_feats  : feats[S*B,24] = hs @ Wout^T + bout (VALU, tiny)
//  6) k_viterbi: forward max-sum + backtrace, tags -> d_out as float

typedef __attribute__((ext_vector_type(16))) _Float16 v16h;
typedef __attribute__((ext_vector_type(8)))  _Float16 v8h;
typedef __attribute__((ext_vector_type(8)))  float    v8f;

// A-matrix 16x32 f16 layout: lane m=lane&15, grp=lane>>4.
// v16h elem i: K = 16*grp + i  -> 16 contiguous halves (matches load_b16).
__device__ __forceinline__ v16h load_a16(const _Float16* rowp_at_kk, int grp) {
  v8h lo = *(const v8h*)(rowp_at_kk + 16 * grp);
  v8h hi = *(const v8h*)(rowp_at_kk + 16 * grp + 8);
  v16h a;
#pragma unroll
  for (int i = 0; i < 8; ++i) { a[i] = lo[i]; a[8 + i] = hi[i]; }
  return a;
}

// B-matrix 32x16 f16 layout: lane holds column n=lane&15; elem i: K = kk + 16*grp + i
// -> 16 consecutive halves (32B, aligned). W stored [N,256] row-major (ld = 256).
__device__ __forceinline__ v16h load_b16(const _Float16* __restrict__ W, int n, int kk, int grp) {
  return *(const v16h*)(W + (size_t)n * 256 + kk + 16 * grp);
}

__device__ __forceinline__ float sigf(float x) { return 1.f / (1.f + __expf(-x)); }

// P is stored as the D-register image of each 16x16 tile:
//   P[((mtile*128 + ntile)*32 + lane)*8 + r]   (v8f per lane, fully coalesced)
__device__ __forceinline__ size_t ptile(int mtile, int ntile, int lane) {
  return ((size_t)(mtile * 128 + ntile) * 32 + lane) * 8;
}

// ---------------- 1) embedding gather + f16 convert ----------------
__global__ void k_gather(const int* __restrict__ sentence, const float* __restrict__ embed,
                         _Float16* __restrict__ X16) {
  int sb = blockIdx.x;              // sb = s*32 + b, 0..8191
  int s = sb >> 5, b = sb & 31;
  int tok = sentence[b * 256 + s];  // sentence is [B,S]
  X16[(size_t)sb * 256 + threadIdx.x] = (_Float16)embed[(size_t)tok * 256 + threadIdx.x];
}

// ---------------- 2) weight f16 conversion + bias sums ----------------
__global__ void k_prep(const float* __restrict__ Wih_f, const float* __restrict__ Wih_b,
                       const float* __restrict__ Whh_f, const float* __restrict__ Whh_b,
                       const float* __restrict__ bih_f, const float* __restrict__ bhh_f,
                       const float* __restrict__ bih_b, const float* __restrict__ bhh_b,
                       _Float16* __restrict__ Wih16, _Float16* __restrict__ Whh16,
                       float* __restrict__ bsum) {
  int id = blockIdx.x * 256 + threadIdx.x;    // 0 .. 2048*256-1
  const int HALF = 1024 * 256;
  Wih16[id] = (_Float16)(id < HALF ? Wih_f[id] : Wih_b[id - HALF]);
  Whh16[id] = (_Float16)(id < HALF ? Whh_f[id] : Whh_b[id - HALF]);
  if (id < 2048)
    bsum[id] = (id < 1024) ? (bih_f[id] + bhh_f[id]) : (bih_b[id - 1024] + bhh_b[id - 1024]);
}

// ---------------- 3) input projection GEMM: P = X16 @ W^T + bsum ----------------
// M=8192 (512 tiles), N=2048 (128 tiles: cols 0..1023 fwd gates, 1024..2047 bwd), K=256.
// Each wave: 1 M-tile x 4 N-tiles (A loaded once per kk, reused 4x).
__global__ __launch_bounds__(256) void k_in_gemm(const _Float16* __restrict__ X,
                                                 const _Float16* __restrict__ W,
                                                 const float* __restrict__ bsum,
                                                 float* __restrict__ P) {
  int lane = threadIdx.x & 31, wave = threadIdx.x >> 5;
  int job = blockIdx.x * 8 + wave;            // 16384 jobs (512 mtiles x 32 n-groups)
  int mtile = job >> 5;                       // 0..511
  int ng = job & 31;                          // 0..31 -> 4 consecutive ntiles
  int grp = lane >> 4, ncol = lane & 15;
  int row0 = mtile * 16;
  v8f acc[4];
#pragma unroll
  for (int t = 0; t < 4; ++t) {
    float bv = bsum[(ng * 4 + t) * 16 + ncol];
#pragma unroll
    for (int r = 0; r < 8; ++r) acc[t][r] = bv;
  }
  const _Float16* arow = X + (size_t)(row0 + (lane & 15)) * 256;
#pragma unroll
  for (int kk = 0; kk < 256; kk += 32) {
    v16h a = load_a16(arow + kk, grp);
#pragma unroll
    for (int t = 0; t < 4; ++t) {
      v16h b = load_b16(W, (ng * 4 + t) * 16 + ncol, kk, grp);
      acc[t] = __builtin_amdgcn_wmma_f32_16x16x32_f16(false, a, false, b, (short)0,
                                                      acc[t], false, false);
    }
  }
#pragma unroll
  for (int t = 0; t < 4; ++t)
    *(v8f*)(P + ptile(mtile, ng * 4 + t, lane)) = acc[t];
}

// ---------------- 4) LSTM recurrence: one workgroup per direction ----------------
// wave -> (mtile = wave&1 : 16 batch rows, htile = wave>>1 : 16 hidden cols).
// Each wave owns i/f/g/o accumulators for its 16x16 (b,h) patch -> in-register gates.
__global__ __launch_bounds__(1024) void k_lstm(const float* __restrict__ P,
                                               const _Float16* __restrict__ Whh16,
                                               float* __restrict__ hs) {
  int dir = blockIdx.x;                       // 0=fwd, 1=bwd
  int tid = threadIdx.x, lane = tid & 31, wave = tid >> 5;
  int mtile = wave & 1, htile = wave >> 1;
  int grp = lane >> 4, ncol = lane & 15;
  __shared__ _Float16 h16[32 * 256];          // h state, f16, [b][h]
  for (int i = tid; i < 32 * 256; i += 1024) h16[i] = (_Float16)0.f;
  float cr[8];
#pragma unroll
  for (int r = 0; r < 8; ++r) cr[r] = 0.f;
  const _Float16* Wd = Whh16 + (size_t)dir * 1024 * 256;
  __syncthreads();

  for (int step = 0; step < 256; ++step) {
    int s = dir ? (255 - step) : step;
    int mtileG = s * 2 + mtile;               // global M-tile of this wave's batch rows
    v8f acc[4];
#pragma unroll
    for (int gi = 0; gi < 4; ++gi)            // coalesced v8f reload of the D-tile image
      acc[gi] = *(const v8f*)(P + ptile(mtileG, dir * 64 + gi * 16 + htile, lane));
    const _Float16* arow = &h16[(mtile * 16 + (lane & 15)) * 256];
#pragma unroll
    for (int kk = 0; kk < 256; kk += 32) {
      v16h a = load_a16(arow + kk, grp);
#pragma unroll
      for (int gi = 0; gi < 4; ++gi) {
        v16h b = load_b16(Wd, gi * 256 + htile * 16 + ncol, kk, grp);
        acc[gi] = __builtin_amdgcn_wmma_f32_16x16x32_f16(false, a, false, b, (short)0,
                                                         acc[gi], false, false);
      }
    }
    __syncthreads();                          // everyone done reading h16
#pragma unroll
    for (int r = 0; r < 8; ++r) {
      int brow = mtile * 16 + r + 8 * grp;
      int hcol = htile * 16 + ncol;
      float ig = sigf(acc[0][r]);
      float fg = sigf(acc[1][r]);
      float gg = tanhf(acc[2][r]);
      float og = sigf(acc[3][r]);
      float c  = fg * cr[r] + ig * gg;
      cr[r] = c;
      float h = og * tanhf(c);
      h16[brow * 256 + hcol] = (_Float16)h;
      hs[((size_t)s * 32 + brow) * 512 + dir * 256 + hcol] = h;  // [S,B,HD], fwd|bwd halves
    }
    __syncthreads();                          // h16 ready for next step
  }
}

// ---------------- 5) output projection ----------------
__global__ void k_feats(const float* __restrict__ hs, const float* __restrict__ Wout,
                        const float* __restrict__ bout, float* __restrict__ feats) {
  int id = blockIdx.x * blockDim.x + threadIdx.x;
  if (id >= 8192 * 24) return;
  int sb = id / 24, t = id % 24;
  const float* h = hs + (size_t)sb * 512;
  const float* w = Wout + (size_t)t * 512;
  float acc = 0.f;
#pragma unroll 8
  for (int k = 0; k < 512; ++k) acc += h[k] * w[k];
  feats[id] = acc + bout[t];
}

// ---------------- 6) Viterbi decode ----------------
__global__ __launch_bounds__(768) void k_viterbi(const float* __restrict__ feats,
                                                 const float* __restrict__ trans,
                                                 const float* __restrict__ start_t,
                                                 const float* __restrict__ stop_t,
                                                 int* __restrict__ vidx,
                                                 float* __restrict__ out) {
  __shared__ float vbuf[2][32 * 24];
  __shared__ float tr[24 * 24];
  int tid = threadIdx.x;                      // tid = b*24 + nt
  if (tid < 576) tr[tid] = trans[tid];
  int b = tid / 24, nt = tid % 24;
  vbuf[0][tid] = feats[b * 24 + nt] + start_t[nt];
  __syncthreads();
  for (int s = 1; s < 256; ++s) {
    const float* vr = vbuf[(s - 1) & 1];
    float* vw = vbuf[s & 1];
    float best = -3.4e38f; int bi = 0;
#pragma unroll
    for (int pt = 0; pt < 24; ++pt) {         // strict '>' keeps first max (jnp.argmax)
      float sc = vr[b * 24 + pt] + tr[pt * 24 + nt];
      if (sc > best) { best = sc; bi = pt; }
    }
    vw[tid] = best + feats[((size_t)s * 32 + b) * 24 + nt];
    vidx[((size_t)(s - 1) * 32 + b) * 24 + nt] = bi;
    __syncthreads();
  }
  if (tid < 32) {                             // backtrace, one thread per batch row
    const float* vf = vbuf[1];                // (s=255) & 1
    float best = -3.4e38f; int tag = 0;
#pragma unroll
    for (int t = 0; t < 24; ++t) {
      float sc = vf[tid * 24 + t] + stop_t[t];
      if (sc > best) { best = sc; tag = t; }
    }
    out[tid * 256 + 255] = (float)tag;
    for (int s = 254; s >= 0; --s) {
      tag = vidx[((size_t)s * 32 + tid) * 24 + tag];
      out[tid * 256 + s] = (float)tag;
    }
  }
}

extern "C" void kernel_launch(void* const* d_in, const int* in_sizes, int n_in,
                              void* d_out, int out_size, void* d_ws, size_t ws_size,
                              hipStream_t stream) {
  (void)in_sizes; (void)n_in; (void)out_size; (void)ws_size;
  const int*   sentence = (const int*)  d_in[0];
  const float* embed    = (const float*)d_in[1];
  const float* Wih_f    = (const float*)d_in[2];
  const float* Whh_f    = (const float*)d_in[3];
  const float* bih_f    = (const float*)d_in[4];
  const float* bhh_f    = (const float*)d_in[5];
  const float* Wih_b    = (const float*)d_in[6];
  const float* Whh_b    = (const float*)d_in[7];
  const float* bih_b    = (const float*)d_in[8];
  const float* bhh_b    = (const float*)d_in[9];
  const float* Wout     = (const float*)d_in[10];
  const float* bout     = (const float*)d_in[11];
  const float* trans    = (const float*)d_in[12];
  const float* start_t  = (const float*)d_in[13];
  const float* stop_t   = (const float*)d_in[14];
  float* out = (float*)d_out;

  char* ws = (char*)d_ws;
  _Float16* X16   = (_Float16*)(ws);                 // 8192*256 f16     = 4,194,304 B
  _Float16* Wih16 = (_Float16*)(ws +  4194304);      // 2048*256 f16     = 1,048,576 B
  _Float16* Whh16 = (_Float16*)(ws +  5242880);      // 2048*256 f16     = 1,048,576 B
  float*    bsum  = (float*)   (ws +  6291456);      // 2048 f32         =     8,192 B
  float*    P     = (float*)   (ws +  6299648);      // 8192*2048 f32    = 67,108,864 B (tile layout)
  float*    hs    = (float*)   (ws + 73408512);      // 8192*512 f32     = 16,777,216 B
  float*    feats = (float*)   (ws + 90185728);      // 8192*24 f32      =   786,432 B
  int*      vidx  = (int*)     (ws + 90972160);      // 255*32*24 i32    =   783,360 B

  k_gather <<<8192, 256, 0, stream>>>(sentence, embed, X16);
  k_prep   <<<2048, 256, 0, stream>>>(Wih_f, Wih_b, Whh_f, Whh_b,
                                      bih_f, bhh_f, bih_b, bhh_b, Wih16, Whh16, bsum);
  k_in_gemm<<<2048, 256, 0, stream>>>(X16, Wih16, bsum, P);
  k_lstm   <<<2, 1024, 0, stream>>>(P, Whh16, hs);
  k_feats  <<<768, 256, 0, stream>>>(hs, Wout, bout, feats);
  k_viterbi<<<1, 768, 0, stream>>>(feats, trans, start_t, stop_t, vidx, out);
}
